// MultiHeadedSelfAttentionModule_38663295599107
// MI455X (gfx1250) — matrix operand order repaired and checked by reading, not compiled, not run on hardware
//
#include <hip/hip_runtime.h>

// ---------------------------------------------------------------------------
// MI455X (gfx1250) Transformer-XL MHSA block.
// wave32; WMMA f32_16x16x32_f16 for all matmuls; scores resident in WGP LDS.
// Weights pre-transposed once so GEMM B-tiles are contiguous global b128 loads
// (L2-resident); attention uses 96KB LDS/block -> 3 workgroups per WGP.
// ---------------------------------------------------------------------------

typedef _Float16 h16_t;
typedef __attribute__((ext_vector_type(16))) _Float16 v16h;
typedef __attribute__((ext_vector_type(8)))  _Float16 v8h;
typedef __attribute__((ext_vector_type(4)))  _Float16 v4h;
typedef __attribute__((ext_vector_type(8)))  float    v8f;

#define T_SEQ 1024
#define DIM   512
#define DH    64

__device__ __forceinline__ v8f wmma_f16(v16h a, v16h b, v8f c) {
  // D = A(16x32 f16) * B(32x16 f16) + C(16x16 f32)
  return __builtin_amdgcn_wmma_f32_16x16x32_f16(false, a, false, b, (short)0, c,
                                                false, false);
}

// A-matrix 16x32 f16 (CDNA5 ISA): lanes 0-15 hold M=0..15 with
// K = {0..7, 16..23}; lanes 16-31 hold same M with K = {8..15, 24..31}.
// -> per lane: two contiguous 16B runs at k0 + 8*half and k0 + 16 + 8*half.
__device__ __forceinline__ v16h load_a(const h16_t* base, int ld, int row0, int k0) {
  int lane = threadIdx.x & 31;
  int hl = lane >> 4, m = lane & 15;
  const h16_t* p = base + (row0 + m) * ld + k0 + hl * 8;
  v8h lo = *(const v8h*)p;
  v8h hi = *(const v8h*)(p + 16);
  v16h a;
#pragma unroll
  for (int j = 0; j < 8; ++j) { a[j] = lo[j]; a[j + 8] = hi[j]; }
  return a;
}

// A tile from the q matrix with per-row clamp (stays inside the batch) and an
// added per-head 64-vector (u or v) held in LDS.
__device__ __forceinline__ v16h load_a_qadd(const h16_t* qb, int row0, int rowmax,
                                            int colbase, int ks, const h16_t* add) {
  int lane = threadIdx.x & 31;
  int hl = lane >> 4, m = lane & 15;
  int r = row0 + m;
  if (r > rowmax) r = rowmax;
  const h16_t* p = qb + r * DIM + colbase + ks + hl * 8;
  v8h lo = *(const v8h*)p;
  v8h hi = *(const v8h*)(p + 16);
  const h16_t* av = add + ks + hl * 8;
  v16h a;
#pragma unroll
  for (int j = 0; j < 8; ++j) { a[j] = lo[j] + av[j]; a[j + 8] = hi[j] + av[j + 16]; }
  return a;
}

// B-matrix 32x16 f16 where B[K][N] = base[(n0+N)*ld + k0 + K] (base holds B^T
// row-major). Layout matches A: lanes 0-15: N=lane, K={0..7,16..23};
// lanes 16-31: N=lane-16, K={8..15,24..31}.
// -> per lane: two contiguous 16B runs at k0 + 8*half and k0 + 16 + 8*half.
__device__ __forceinline__ v16h load_bt(const h16_t* base, int ld, int n0, int k0) {
  int lane = threadIdx.x & 31;
  int hl = lane >> 4, n = lane & 15;
  const h16_t* p = base + (n0 + n) * ld + k0 + hl * 8;
  v8h lo = *(const v8h*)p;
  v8h hi = *(const v8h*)(p + 16);
  v16h b;
#pragma unroll
  for (int j = 0; j < 8; ++j) { b[j] = lo[j]; b[j + 8] = hi[j]; }
  return b;
}

// ---------------------------------------------------------------------------
// LayerNorm(512) -> f16.  One block per row, 128 threads, 4 elems/thread.
// ---------------------------------------------------------------------------
__global__ void __launch_bounds__(128)
ln512(const float* __restrict__ x, const float* __restrict__ gamma,
      const float* __restrict__ beta, h16_t* __restrict__ out) {
  __shared__ float red[8];
  const int row = blockIdx.x;
  const int tid = threadIdx.x, lane = tid & 31, wave = tid >> 5;
  const int c0 = tid * 4;
  float4 v4 = *(const float4*)(x + row * DIM + c0);
  float s  = v4.x + v4.y + v4.z + v4.w;
  float ss = v4.x * v4.x + v4.y * v4.y + v4.z * v4.z + v4.w * v4.w;
#pragma unroll
  for (int m = 16; m >= 1; m >>= 1) {
    s  += __shfl_xor(s, m, 32);
    ss += __shfl_xor(ss, m, 32);
  }
  if (lane == 0) { red[wave] = s; red[4 + wave] = ss; }
  __syncthreads();
  float ts  = red[0] + red[1] + red[2] + red[3];
  float tss = red[4] + red[5] + red[6] + red[7];
  float mu  = ts * (1.0f / 512.0f);
  float var = tss * (1.0f / 512.0f) - mu * mu;
  float rs  = rsqrtf(var + 1e-3f);
  v4h o;
  o[0] = (h16_t)((v4.x - mu) * rs * gamma[c0 + 0] + beta[c0 + 0]);
  o[1] = (h16_t)((v4.y - mu) * rs * gamma[c0 + 1] + beta[c0 + 1]);
  o[2] = (h16_t)((v4.z - mu) * rs * gamma[c0 + 2] + beta[c0 + 2]);
  o[3] = (h16_t)((v4.w - mu) * rs * gamma[c0 + 3] + beta[c0 + 3]);
  *(v4h*)(out + row * DIM + c0) = o;
}

// f32 -> f16 convert, 4 elems/thread (activations: pe)
__global__ void cvt4(const float* __restrict__ src, h16_t* __restrict__ dst, int n4) {
  int i = blockIdx.x * blockDim.x + threadIdx.x;
  if (i < n4) {
    float4 f = ((const float4*)src)[i];
    v4h o;
    o[0] = (h16_t)f.x; o[1] = (h16_t)f.y; o[2] = (h16_t)f.z; o[3] = (h16_t)f.w;
    ((v4h*)dst)[i] = o;
  }
}

// f32 [512][512] weight -> f16 transposed [N][K] (once per call; L2-resident
// afterwards, so GEMM B-tiles become contiguous b128 loads with no LDS pass).
__global__ void __launch_bounds__(256)
cvtT512(const float* __restrict__ src, h16_t* __restrict__ dst) {
  int idx = blockIdx.x * 256 + threadIdx.x;   // 262144 threads
  int k = idx >> 9, n = idx & 511;
  dst[n * 512 + k] = (h16_t)src[idx];
}

// ---------------------------------------------------------------------------
// C[M=8192, N=512] = A[8192x512]f16 @ W[512x512] (+bias)(+residual)
// WT = W^T f16 [N][K].  Block: 128 thr = 4 waves; wave tile 32x64 (2x4 WMMA),
// block tile 64x128.  No LDS, no barriers: pure global b128 + WMMA clauses.
// ---------------------------------------------------------------------------
__global__ void __launch_bounds__(128)
gemm512(const h16_t* __restrict__ A, const h16_t* __restrict__ WT,
        const float* __restrict__ bias, const float* __restrict__ residual,
        h16_t* __restrict__ out16, float* __restrict__ out32) {
  const int tid = threadIdx.x;
  const int wave = tid >> 5, lane = tid & 31;
  const int hl = lane >> 4, ln = lane & 15;
  const int wm = wave >> 1, wn = wave & 1;
  const int m0 = blockIdx.x * 64 + wm * 32;
  const int n0 = blockIdx.y * 128 + wn * 64;
  v8f acc[2][4] = {};
  for (int ks = 0; ks < 512; ks += 32) {
    v16h a0 = load_a(A, 512, m0, ks);
    v16h a1 = load_a(A, 512, m0 + 16, ks);
    v16h b0 = load_bt(WT, 512, n0, ks);
    v16h b1 = load_bt(WT, 512, n0 + 16, ks);
    v16h b2 = load_bt(WT, 512, n0 + 32, ks);
    v16h b3 = load_bt(WT, 512, n0 + 48, ks);
    acc[0][0] = wmma_f16(a0, b0, acc[0][0]);
    acc[0][1] = wmma_f16(a0, b1, acc[0][1]);
    acc[0][2] = wmma_f16(a0, b2, acc[0][2]);
    acc[0][3] = wmma_f16(a0, b3, acc[0][3]);
    acc[1][0] = wmma_f16(a1, b0, acc[1][0]);
    acc[1][1] = wmma_f16(a1, b1, acc[1][1]);
    acc[1][2] = wmma_f16(a1, b2, acc[1][2]);
    acc[1][3] = wmma_f16(a1, b3, acc[1][3]);
  }
#pragma unroll
  for (int mt = 0; mt < 2; ++mt)
#pragma unroll
    for (int nt = 0; nt < 4; ++nt)
#pragma unroll
      for (int r = 0; r < 8; ++r) {
        int row = m0 + mt * 16 + r + 8 * hl;   // C layout: M = r + 8*half
        int col = n0 + nt * 16 + ln;           //           N = lane&15
        float val = acc[mt][nt][r];
        if (bias) val += bias[col];
        if (out32) {
          if (residual) val += residual[row * 512 + col];
          out32[row * 512 + col] = val;
        } else {
          out16[row * 512 + col] = (h16_t)val;
        }
      }
}

// ---------------------------------------------------------------------------
// Transformer-XL attention: one block = one (b,h) and 16 query rows.
// Dynamic LDS (96KB -> 3 workgroups/WGP):
//   Xs f16 [32][1024] = 64KB   positional strip; later aliased by Vt[64][512]
//   S  f16 [16][1024] = 32KB   score strip; softmax'd in place -> P
// ---------------------------------------------------------------------------
__global__ void __launch_bounds__(128)
attn_xl(const h16_t* __restrict__ q16, const h16_t* __restrict__ k16,
        const h16_t* __restrict__ v16m, const h16_t* __restrict__ p16,
        const float* __restrict__ uvec, const float* __restrict__ vvec,
        h16_t* __restrict__ o16) {
  extern __shared__ char smem[];
  h16_t* Xs = (h16_t*)smem;                    // 32*1024 halfs = 65536 B
  h16_t* S  = (h16_t*)(smem + 65536);          // 16*1024 halfs = 32768 B
  h16_t* P  = S;                               // softmax in place
  h16_t* Vt = Xs;                              // 64*512 halfs, alias (Xs dead)
  __shared__ h16_t uvl[64], vvl[64];

  const int tid = threadIdx.x;
  const int wave = tid >> 5, lane = tid & 31;
  const int hl = lane >> 4, ln = lane & 15;
  const int t0 = blockIdx.x * 16;
  const int b  = blockIdx.y >> 3;
  const int hd = blockIdx.y & 7;
  const int bT = b * T_SEQ;
  const int colbase = hd * DH;

  if (tid < 64) {
    uvl[tid] = (h16_t)uvec[colbase + tid];
    vvl[tid] = (h16_t)vvec[colbase + tid];
  }
  __syncthreads();

  // ---- Phase 1: Xs[qq][r] = (q[t0+qq] + v_h) . p[r]   (qq<32, r<1024) ----
  {
    int mt = wave & 1;                    // wave -> fixed row-tile, nt parity
    int row0 = bT + t0 + mt * 16;
    v16h a0 = load_a_qadd(q16, row0, bT + T_SEQ - 1, colbase, 0, vvl);
    v16h a1 = load_a_qadd(q16, row0, bT + T_SEQ - 1, colbase, 32, vvl);
    const h16_t* pb = p16 + bT * DIM;
    for (int nt = (wave >> 1); nt < 64; nt += 2) {
      v8f c = {};
      v16h b0 = load_bt(pb, DIM, nt * 16, colbase + 0);
      v16h b1 = load_bt(pb, DIM, nt * 16, colbase + 32);
      c = wmma_f16(a0, b0, c);
      c = wmma_f16(a1, b1, c);
      int Mb = mt * 16 + 8 * hl;
      int Nc = nt * 16 + ln;
#pragma unroll
      for (int r = 0; r < 8; ++r) Xs[(Mb + r) * 1024 + Nc] = (h16_t)c[r];
    }
  }
  __syncthreads();

  // ---- Phase 2: S = ((q+u).k^T + rel_shift(Xs)) * 1/sqrt(64) ----
  {
    v16h a0 = load_a_qadd(q16, bT + t0, bT + T_SEQ - 1, colbase, 0, uvl);
    v16h a1 = load_a_qadd(q16, bT + t0, bT + T_SEQ - 1, colbase, 32, uvl);
    const h16_t* kb = k16 + bT * DIM;
    for (int nt = wave; nt < 64; nt += 4) {
      v8f c = {};
      v16h b0 = load_bt(kb, DIM, nt * 16, colbase + 0);
      v16h b1 = load_bt(kb, DIM, nt * 16, colbase + 32);
      c = wmma_f16(a0, b0, c);
      c = wmma_f16(a1, b1, c);
#pragma unroll
      for (int r = 0; r < 8; ++r) {
        int M = r + 8 * hl;
        int t = t0 + M;
        int kk = nt * 16 + ln;
        // rel_shift: out[q,k] = X[q, T-1+k-q] (k<=q); 0 (k==q+1);
        //            X[q+1, k-q-2] (k>q+1)
        float bd;
        if (kk <= t)          bd = (float)Xs[M * 1024 + (kk - t + (T_SEQ - 1))];
        else if (kk == t + 1) bd = 0.0f;
        else                  bd = (float)Xs[(M + 1) * 1024 + (kk - t - 2)];
        S[M * 1024 + kk] = (h16_t)((c[r] + bd) * 0.125f);
      }
    }
  }
  __syncthreads();

  // ---- Phase 3a: stage V^T tokens [0,512) into Vt (Xs now dead) ----
  for (int ch = tid; ch < 4096; ch += 128) {
    int tok = ch >> 3;
    int d0 = (ch & 7) * 8;
    v8h g = *(const v8h*)(v16m + (bT + tok) * DIM + colbase + d0);
#pragma unroll
    for (int j = 0; j < 8; ++j) Vt[(d0 + j) * 512 + tok] = g[j];
  }
  // ---- Phase 3b: softmax over each of 16 rows (4 rows/wave), in place ----
  for (int rr = wave; rr < 16; rr += 4) {
    h16_t* Sr = S + rr * 1024;
    float mx = -3.4e38f;
    for (int ci = lane; ci < 1024; ci += 32) mx = fmaxf(mx, (float)Sr[ci]);
#pragma unroll
    for (int m = 16; m >= 1; m >>= 1) mx = fmaxf(mx, __shfl_xor(mx, m, 32));
    float sum = 0.f;
    for (int ci = lane; ci < 1024; ci += 32) {
      float e = __expf((float)Sr[ci] - mx);
      Sr[ci] = (h16_t)e;
      sum += e;
    }
#pragma unroll
    for (int m = 16; m >= 1; m >>= 1) sum += __shfl_xor(sum, m, 32);
    float inv = 1.0f / sum;
    for (int ci = lane; ci < 1024; ci += 32)
      Sr[ci] = (h16_t)((float)Sr[ci] * inv);
  }
  __syncthreads();

  // ---- Phase 4: O[16x64] = P[16x1024] @ V[1024x64], one 16-col tile/wave,
  //      V^T staged in two 512-token halves (Vt region is 64KB) ----
  {
    int n0 = wave * 16;
    v8f c = {};
    for (int k0 = 0; k0 < 512; k0 += 32) {
      v16h a  = load_a(P, 1024, 0, k0);
      v16h bb = load_bt(Vt, 512, n0, k0);
      c = wmma_f16(a, bb, c);
    }
    __syncthreads();
    for (int ch = tid; ch < 4096; ch += 128) {   // stage tokens [512,1024)
      int tok = ch >> 3;
      int d0 = (ch & 7) * 8;
      v8h g = *(const v8h*)(v16m + (bT + 512 + tok) * DIM + colbase + d0);
#pragma unroll
      for (int j = 0; j < 8; ++j) Vt[(d0 + j) * 512 + tok] = g[j];
    }
    __syncthreads();
    for (int k0 = 512; k0 < 1024; k0 += 32) {
      v16h a  = load_a(P, 1024, 0, k0);
      v16h bb = load_bt(Vt, 512, n0, k0 - 512);
      c = wmma_f16(a, bb, c);
    }
#pragma unroll
    for (int r = 0; r < 8; ++r)
      o16[(bT + t0 + r + 8 * hl) * DIM + colbase + n0 + ln] = (h16_t)c[r];
  }
}

// ---------------------------------------------------------------------------
extern "C" void kernel_launch(void* const* d_in, const int* in_sizes, int n_in,
                              void* d_out, int out_size, void* d_ws, size_t ws_size,
                              hipStream_t stream) {
  (void)in_sizes; (void)n_in; (void)out_size; (void)ws_size;
  const float* x     = (const float*)d_in[0];
  const float* pe    = (const float*)d_in[1];
  const float* gamma = (const float*)d_in[2];
  const float* beta  = (const float*)d_in[3];
  const float* wq    = (const float*)d_in[4];
  const float* bq    = (const float*)d_in[5];
  const float* wk    = (const float*)d_in[6];
  const float* bk    = (const float*)d_in[7];
  const float* wv    = (const float*)d_in[8];
  const float* bv    = (const float*)d_in[9];
  const float* wpos  = (const float*)d_in[10];
  const float* u     = (const float*)d_in[11];
  const float* vvec  = (const float*)d_in[12];
  const float* wout  = (const float*)d_in[13];
  const float* bout  = (const float*)d_in[14];
  float* out = (float*)d_out;

  const int NT  = 8192 * 512;   // 4194304 elems (B*T x D)
  const int WSZ = 512 * 512;    // 262144 elems
  h16_t* ws    = (h16_t*)d_ws;
  h16_t* h16   = ws;
  h16_t* pe16  = h16 + NT;
  h16_t* wq16t = pe16 + NT;     // all weights stored transposed [N][K] f16
  h16_t* wk16t = wq16t + WSZ;
  h16_t* wv16t = wk16t + WSZ;
  h16_t* wp16t = wv16t + WSZ;
  h16_t* wo16t = wp16t + WSZ;
  h16_t* q16   = wo16t + WSZ;
  h16_t* k16   = q16 + NT;
  h16_t* v16m  = k16 + NT;
  h16_t* p16   = v16m + NT;
  h16_t* o16   = p16 + NT;

  // LayerNorm + precision converts (+ one-time weight transpose)
  ln512<<<8192, 128, 0, stream>>>(x, gamma, beta, h16);
  cvt4<<<(NT / 4 + 255) / 256, 256, 0, stream>>>(pe, pe16, NT / 4);
  cvtT512<<<WSZ / 256, 256, 0, stream>>>(wq, wq16t);
  cvtT512<<<WSZ / 256, 256, 0, stream>>>(wk, wk16t);
  cvtT512<<<WSZ / 256, 256, 0, stream>>>(wv, wv16t);
  cvtT512<<<WSZ / 256, 256, 0, stream>>>(wpos, wp16t);
  cvtT512<<<WSZ / 256, 256, 0, stream>>>(wout, wo16t);

  // q/k/v/p projections (WMMA GEMMs, LDS-free)
  dim3 gg(128, 4);   // 64-row x 128-col tiles
  gemm512<<<gg, 128, 0, stream>>>(h16,  wq16t, bq, nullptr, q16, nullptr);
  gemm512<<<gg, 128, 0, stream>>>(h16,  wk16t, bk, nullptr, k16, nullptr);
  gemm512<<<gg, 128, 0, stream>>>(h16,  wv16t, bv, nullptr, v16m, nullptr);
  gemm512<<<gg, 128, 0, stream>>>(pe16, wp16t, nullptr, nullptr, p16, nullptr);

  // attention with fused rel_shift + softmax, scores resident in 96KB LDS
  const int SMEM = 32 * 1024 * 2 + 16 * 1024 * 2;  // 98304 B
  hipFuncSetAttribute(reinterpret_cast<const void*>(attn_xl),
                      hipFuncAttributeMaxDynamicSharedMemorySize, SMEM);
  attn_xl<<<dim3(64, 64), 128, SMEM, stream>>>(q16, k16, v16m, p16, u, vvec, o16);

  // output projection + bias + residual (f32 out)
  gemm512<<<gg, 128, 0, stream>>>(o16, wo16t, bout, x, nullptr, out);
}
